// LSTM_GNN_60902636257637
// MI455X (gfx1250) — compile-verified, matrix operand order checked by reading, not run on hardware
//
#include <hip/hip_runtime.h>
#include <math.h>

// ---------------- CDNA5 WMMA types ----------------
typedef __attribute__((ext_vector_type(16))) _Float16 v16h;
typedef __attribute__((ext_vector_type(8)))  _Float16 v8h;
typedef __attribute__((ext_vector_type(8)))  float    v8f;

#define NWG 16
#define BLK 256
#define Hh  256          // hidden
#define Bb  64           // batch
#define Tt  512          // time

__device__ __forceinline__ float sigf(float x){ return 1.0f/(1.0f+__expf(-x)); }

__device__ __forceinline__ v8f wmma16(v16h a, v16h b, v8f c){
  // D = A(16x32 f16) * B(32x16 f16) + C(16x16 f32)
  return __builtin_amdgcn_wmma_f32_16x16x32_f16(false, a, false, b, (short)0, c, false, false);
}

// A-fragment (ISA 7.12.2, 16-bit A 16x32): lane L holds row M=L%16;
// elements 0..7 = K kb..kb+7, elements 8..15 = K kb+16..kb+23,
// where kb = kc*32 + (L<16 ? 0 : 8).
__device__ __forceinline__ v16h load_A(const _Float16* __restrict__ base, int ld,
                                       int row, int kb){
  const _Float16* p = base + (size_t)row*ld + kb;
  union { v16h v; v8h h2[2]; } A;
  A.h2[0] = *(const v8h*)(p);
  A.h2[1] = *(const v8h*)(p + 16);
  return A.v;
}

// ---------------- weight prep: f32 -> f16, swizzled B-fragments ----------------
// B-fragment (dense 32x16, by analogy w/ sparse table): lane L holds col N=L%16,
// elements e=0..15 hold K = kc*32 + kmap(L,e) (table order)... stored pre-swizzled with
// the SAME (lane,e)->k map as A so prep below writes whatever map we pick;
// we use the A-style interleave for both so GEMM indexing is consistent.
__device__ __forceinline__ int kmap(int lane, int e){
  return (e < 8) ? (((lane < 16) ? 0 : 8) + e)
                 : (((lane < 16) ? 16 : 24) + (e - 8));
}

__global__ void __launch_bounds__(256) prep_kernel(
    const float* __restrict__ Whh0, const float* __restrict__ Wih1,
    const float* __restrict__ Whh1,
    const float* __restrict__ bih0, const float* __restrict__ bhh0,
    const float* __restrict__ bih1, const float* __restrict__ bhh1,
    const float* __restrict__ Wr1,
    _Float16* __restrict__ frag0,   // [64 nt][8 kc][32 lane][16 e]
    _Float16* __restrict__ frag1,   // [64 nt][16 kc][32 lane][16 e]
    float* __restrict__ bias0, float* __restrict__ bias1,
    _Float16* __restrict__ wr1h,
    _Float16* __restrict__ h0buf, _Float16* __restrict__ h1buf)
{
  const int F0 = 64*8*32*16;      // 262144
  const int F1 = 64*16*32*16;     // 524288
  const int NB = 1024;
  const int NW = 100*256;
  const int NZ = 2*2*Bb*Hh;       // zero both double-buffered h states
  const int total = F0 + F1 + 2*NB + NW + NZ;
  for (int idx = blockIdx.x*blockDim.x + threadIdx.x; idx < total;
       idx += gridDim.x*blockDim.x){
    int i = idx;
    if (i < F0){
      int e = i & 15, lane = (i >> 4) & 31, kc = (i >> 9) & 7, nt = i >> 12;
      int k = kc*32 + kmap(lane, e);
      int n = nt*16 + (lane & 15);
      frag0[i] = (_Float16)Whh0[(size_t)n*Hh + k];
      continue;
    }
    i -= F0;
    if (i < F1){
      int e = i & 15, lane = (i >> 4) & 31, kc = (i >> 9) & 15, nt = i >> 13;
      int k = kc*32 + kmap(lane, e);
      int n = nt*16 + (lane & 15);
      float v = (k < Hh) ? Wih1[(size_t)n*Hh + k] : Whh1[(size_t)n*Hh + (k - Hh)];
      frag1[i] = (_Float16)v;
      continue;
    }
    i -= F1;
    if (i < NB){ bias0[i] = bih0[i] + bhh0[i]; continue; }
    i -= NB;
    if (i < NB){ bias1[i] = bih1[i] + bhh1[i]; continue; }
    i -= NB;
    if (i < NW){ wr1h[i] = (_Float16)Wr1[i]; continue; }
    i -= NW;
    if (i < NZ/2) h0buf[i] = (_Float16)0.f;
    else          h1buf[i - NZ/2] = (_Float16)0.f;
  }
}

// ---------------- persistent 2-layer LSTM recurrence ----------------
__global__ void __launch_bounds__(BLK) lstm_persist(
    const float* __restrict__ x,      // (64,512,8)
    const float* __restrict__ Wih0,   // (1024,8)
    const float* __restrict__ bias0,  // 1024
    const float* __restrict__ bias1,  // 1024
    const _Float16* __restrict__ frag0,
    const _Float16* __restrict__ frag1,
    _Float16* __restrict__ h0buf,     // 2 x (64x256)
    _Float16* __restrict__ h1buf,     // 2 x (64x256)
    _Float16* __restrict__ hist,      // (512,64,256) layer-2 h history
    float* __restrict__ hmean,        // (64,256) time-mean of layer-2 h
    unsigned* __restrict__ bar)
{
  __shared__ float lds_g[4][Bb][16];  // gate tiles i/f/g/o for this WG's 16 cols
  __shared__ float lds_c0[Bb][16];
  __shared__ float lds_c1[Bb][16];
  __shared__ float lds_hs[Bb][16];

  const int tid   = threadIdx.x;
  const int lane  = tid & 31;
  const int wid   = tid >> 5;          // 8 waves
  const int mtile = wid & 3;           // batch tile (4 x 16 rows)
  const int pair  = wid >> 2;          // 0 -> gates {i,f}, 1 -> gates {g,o}
  const int g0    = pair*2, g1 = g0 + 1;
  const int wg    = blockIdx.x;        // 16 WGs, each owns 16 hidden cols
  const int col0  = wg*16;
  const int nt0   = g0*16 + wg;        // global 16-col tile index of gate g0
  const int nt1   = g1*16 + wg;

  for (int i = tid; i < Bb*16; i += BLK){
    ((float*)lds_c0)[i] = 0.f; ((float*)lds_c1)[i] = 0.f; ((float*)lds_hs)[i] = 0.f;
  }
  __syncthreads();

  const int arow = mtile*16 + (lane & 15);
  const int koff = (lane < 16) ? 0 : 8;
  const size_t lstride = (size_t)32*16;   // halves per (nt,kc) fragment block
  unsigned phase = 0;

  for (int t = 0; t < Tt; ++t){
    const int par = t & 1;
    const _Float16* h0c = h0buf + (size_t)par*(Bb*Hh);
    _Float16*       h0n = h0buf + (size_t)(par^1)*(Bb*Hh);
    const _Float16* h1c = h1buf + (size_t)par*(Bb*Hh);
    _Float16*       h1n = h1buf + (size_t)(par^1)*(Bb*Hh);

    // ======== layer 0: gates = h0_{t-1} @ Whh0^T  (K = 256) ========
    {
      v8f acc0 = {}; v8f acc1 = {};
      #pragma unroll 2
      for (int kc = 0; kc < 8; ++kc){
        v16h a  = load_A(h0c, Hh, arow, kc*32 + koff);
        v16h b0 = *(const v16h*)(frag0 + ((size_t)(nt0*8 + kc))*lstride + (size_t)lane*16);
        v16h b1 = *(const v16h*)(frag0 + ((size_t)(nt1*8 + kc))*lstride + (size_t)lane*16);
        acc0 = wmma16(a, b0, acc0);
        acc1 = wmma16(a, b1, acc1);
      }
      union { v8f v; float f[8]; } D0, D1; D0.v = acc0; D1.v = acc1;
      const int n = lane & 15;
      const int rbase = mtile*16 + ((lane < 16) ? 0 : 8);
      #pragma unroll
      for (int v = 0; v < 8; ++v){
        lds_g[g0][rbase + v][n] = D0.f[v];
        lds_g[g1][rbase + v][n] = D1.f[v];
      }
    }
    __syncthreads();

    // elementwise layer 0 (adds x_t @ Wih0^T with K=8 in VALU) -> h0_t
    for (int idx = tid; idx < Bb*16; idx += BLK){
      const int b = idx >> 4, n = idx & 15, col = col0 + n;
      const float* xr = x + ((size_t)b*Tt + t)*8;
      float gv[4];
      #pragma unroll
      for (int g = 0; g < 4; ++g){
        const int r = g*Hh + col;
        float s = lds_g[g][b][n] + bias0[r];
        const float* wr = Wih0 + (size_t)r*8;
        #pragma unroll
        for (int k = 0; k < 8; ++k) s += xr[k]*wr[k];
        gv[g] = s;
      }
      float c = sigf(gv[1])*lds_c0[b][n] + sigf(gv[0])*tanhf(gv[2]);
      lds_c0[b][n] = c;
      float h = sigf(gv[3])*tanhf(c);
      h0n[(size_t)b*Hh + col] = (_Float16)h;
    }

    // ---- grid barrier #1: publish h0_t ----
    __threadfence();
    __syncthreads();
    if (tid == 0){
      ++phase;
      __hip_atomic_fetch_add(bar, 1u, __ATOMIC_RELEASE, __HIP_MEMORY_SCOPE_AGENT);
      while (__hip_atomic_load(bar, __ATOMIC_ACQUIRE, __HIP_MEMORY_SCOPE_AGENT)
             < phase*(unsigned)NWG)
        __builtin_amdgcn_s_sleep(2);
    }
    __syncthreads();
    __threadfence();

    // ======== layer 1: gates = [h0_t ; h1_{t-1}] @ [Wih1;Whh1]^T (K = 512) ====
    {
      v8f acc0 = {}; v8f acc1 = {};
      #pragma unroll 2
      for (int kc = 0; kc < 16; ++kc){
        const _Float16* src = (kc < 8) ? h0n : h1c;
        const int kb = (kc < 8) ? (kc*32 + koff) : (kc*32 + koff - Hh);
        v16h a  = load_A(src, Hh, arow, kb);
        const _Float16* pb0 = frag1 + ((size_t)(nt0*16 + kc))*lstride + (size_t)lane*16;
        const _Float16* pb1 = frag1 + ((size_t)(nt1*16 + kc))*lstride + (size_t)lane*16;
        if (kc < 15){  // prefetch next chunk's B fragments (global_prefetch)
          __builtin_prefetch(pb0 + lstride*1, 0, 1);
          __builtin_prefetch(pb1 + lstride*1, 0, 1);
        }
        v16h b0 = *(const v16h*)pb0;
        v16h b1 = *(const v16h*)pb1;
        acc0 = wmma16(a, b0, acc0);
        acc1 = wmma16(a, b1, acc1);
      }
      union { v8f v; float f[8]; } D0, D1; D0.v = acc0; D1.v = acc1;
      const int n = lane & 15;
      const int rbase = mtile*16 + ((lane < 16) ? 0 : 8);
      #pragma unroll
      for (int v = 0; v < 8; ++v){
        lds_g[g0][rbase + v][n] = D0.f[v];
        lds_g[g1][rbase + v][n] = D1.f[v];
      }
    }
    __syncthreads();

    // elementwise layer 1 -> h1_t, history, running mean
    for (int idx = tid; idx < Bb*16; idx += BLK){
      const int b = idx >> 4, n = idx & 15, col = col0 + n;
      float gv[4];
      #pragma unroll
      for (int g = 0; g < 4; ++g)
        gv[g] = lds_g[g][b][n] + bias1[g*Hh + col];
      float c = sigf(gv[1])*lds_c1[b][n] + sigf(gv[0])*tanhf(gv[2]);
      lds_c1[b][n] = c;
      float h = sigf(gv[3])*tanhf(c);
      _Float16 hf = (_Float16)h;
      h1n[(size_t)b*Hh + col] = hf;
      hist[((size_t)t*Bb + b)*Hh + col] = hf;
      lds_hs[b][n] += h;
    }

    // ---- grid barrier #2: publish h1_t ----
    __threadfence();
    __syncthreads();
    if (tid == 0){
      ++phase;
      __hip_atomic_fetch_add(bar, 1u, __ATOMIC_RELEASE, __HIP_MEMORY_SCOPE_AGENT);
      while (__hip_atomic_load(bar, __ATOMIC_ACQUIRE, __HIP_MEMORY_SCOPE_AGENT)
             < phase*(unsigned)NWG)
        __builtin_amdgcn_s_sleep(2);
    }
    __syncthreads();
    __threadfence();
  }

  for (int idx = tid; idx < Bb*16; idx += BLK){
    const int b = idx >> 4, n = idx & 15;
    hmean[(size_t)b*Hh + col0 + n] = lds_hs[b][n] * (1.0f/512.0f);
  }
}

// ---------------- head kernel 1: m1 and elu(t=511) ----------------
__global__ void __launch_bounds__(128) head1_kernel(
    const _Float16* __restrict__ hist, const float* __restrict__ hmean,
    const float* __restrict__ Wl1, const float* __restrict__ b1v,
    const _Float16* __restrict__ wr1h,
    float* __restrict__ m1, float* __restrict__ elu_last)
{
  __shared__ _Float16 wsh[100*256];   // 50 KB
  __shared__ float hsh[256];
  __shared__ float hm[256];
  const int b = blockIdx.x, tid = threadIdx.x;
  for (int i = tid; i < 100*256; i += 128) wsh[i] = wr1h[i];
  for (int i = tid; i < 256; i += 128) hm[i] = hmean[(size_t)b*256 + i];
  __syncthreads();

  float mterm = 0.f;
  if (tid < 100){
    const float* w = Wl1 + (size_t)tid*256;
    float s = b1v[tid];
    for (int k = 0; k < 256; ++k) s += hm[k]*w[k];
    mterm = s;
  }
  float acc = 0.f, el = 0.f;
  for (int t = 0; t < Tt; ++t){
    __syncthreads();
    const _Float16* hr = hist + ((size_t)t*Bb + b)*256;
    for (int i = tid; i < 256; i += 128) hsh[i] = (float)hr[i];
    __syncthreads();
    if (tid < 100){
      const _Float16* w = wsh + (size_t)tid*256;
      float s = mterm;
      for (int k = 0; k < 256; ++k) s += hsh[k]*(float)w[k];
      float e = (s > 0.f) ? s : (__expf(s) - 1.0f);
      acc += e;
      if (t == Tt-1) el = e;
    }
  }
  if (tid < 100){
    m1[b*100 + tid]       = acc*(1.0f/512.0f);
    elu_last[b*100 + tid] = el;
  }
}

// ---------------- head kernel 2: s = m1@Wl2^T + b2 + elu_last@Wr2^T ----------
__global__ void __launch_bounds__(128) head2_kernel(
    const float* __restrict__ m1, const float* __restrict__ elu_last,
    const float* __restrict__ Wl2, const float* __restrict__ b2v,
    const float* __restrict__ Wr2, float* __restrict__ sbuf)
{
  __shared__ float m[100], e[100];
  const int b = blockIdx.x, j = threadIdx.x;
  if (j < 100){ m[j] = m1[b*100 + j]; e[j] = elu_last[b*100 + j]; }
  __syncthreads();
  const float* wl = Wl2 + (size_t)j*100;
  const float* wr = Wr2 + (size_t)j*100;
  float s = b2v[j];
  for (int k = 0; k < 100; ++k) s += m[k]*wl[k] + e[k]*wr[k];
  sbuf[b*128 + j] = s;
}

// ---------------- head kernel 3: per-k MLP -> pred (8,64,24) ----------------
__global__ void __launch_bounds__(64) head3_kernel(
    const float* __restrict__ sbuf,
    const float* __restrict__ Wfc1, const float* __restrict__ bfc1,
    const float* __restrict__ Wfc2, const float* __restrict__ bfc2,
    float* __restrict__ out)
{
  const int k = blockIdx.x >> 6;
  const int b = blockIdx.x & 63;
  const int tid = threadIdx.x;
  __shared__ float ss[128], z[64];
  ss[tid]      = sbuf[b*128 + tid];
  ss[tid + 64] = sbuf[b*128 + tid + 64];
  __syncthreads();
  {
    const float* w = Wfc1 + ((size_t)k*64 + tid)*128;
    float s = bfc1[k*64 + tid];
    for (int d = 0; d < 128; ++d) s += ss[d]*w[d];
    z[tid] = (s > 0.f) ? s : 0.f;
  }
  __syncthreads();
  if (tid < 24){
    const float* w = Wfc2 + ((size_t)k*24 + tid)*64;
    float s = bfc2[k*24 + tid];
    for (int o = 0; o < 64; ++o) s += z[o]*w[o];
    out[((size_t)k*64 + b)*24 + tid] = s;
  }
}

// ---------------- launch ----------------
static inline size_t alignup(size_t x){ return (x + 255) & ~(size_t)255; }

extern "C" void kernel_launch(void* const* d_in, const int* in_sizes, int n_in,
                              void* d_out, int out_size, void* d_ws, size_t ws_size,
                              hipStream_t stream) {
  const float* x    = (const float*)d_in[0];
  const float* Wih0 = (const float*)d_in[1];
  const float* Whh0 = (const float*)d_in[2];
  const float* bih0 = (const float*)d_in[3];
  const float* bhh0 = (const float*)d_in[4];
  const float* Wih1 = (const float*)d_in[5];
  const float* Whh1 = (const float*)d_in[6];
  const float* bih1 = (const float*)d_in[7];
  const float* bhh1 = (const float*)d_in[8];
  const float* Wl1  = (const float*)d_in[9];
  const float* Wr1  = (const float*)d_in[10];
  const float* b1v  = (const float*)d_in[11];
  const float* Wl2  = (const float*)d_in[12];
  const float* Wr2  = (const float*)d_in[13];
  const float* b2v  = (const float*)d_in[14];
  const float* Wfc1 = (const float*)d_in[15];
  const float* bfc1 = (const float*)d_in[16];
  const float* Wfc2 = (const float*)d_in[17];
  const float* bfc2 = (const float*)d_in[18];
  float* out = (float*)d_out;

  char* ws = (char*)d_ws;
  size_t off = 0;
  unsigned*  bar   = (unsigned*) (ws + off); off = alignup(off + 256);
  _Float16*  h0buf = (_Float16*) (ws + off); off = alignup(off + 2*64*256*2);
  _Float16*  h1buf = (_Float16*) (ws + off); off = alignup(off + 2*64*256*2);
  float*     hmean = (float*)    (ws + off); off = alignup(off + 64*256*4);
  float*     bias0 = (float*)    (ws + off); off = alignup(off + 1024*4);
  float*     bias1 = (float*)    (ws + off); off = alignup(off + 1024*4);
  _Float16*  wr1h  = (_Float16*) (ws + off); off = alignup(off + 100*256*2);
  float*     m1    = (float*)    (ws + off); off = alignup(off + 64*100*4);
  float*     elu_l = (float*)    (ws + off); off = alignup(off + 64*100*4);
  float*     sbuf  = (float*)    (ws + off); off = alignup(off + 64*128*4);
  _Float16*  frag0 = (_Float16*) (ws + off); off = alignup(off + (size_t)64*8*32*16*2);
  _Float16*  frag1 = (_Float16*) (ws + off); off = alignup(off + (size_t)64*16*32*16*2);
  _Float16*  hist  = (_Float16*) (ws + off); off = alignup(off + (size_t)512*64*256*2);

  hipMemsetAsync(bar, 0, 256, stream);

  prep_kernel<<<1024, 256, 0, stream>>>(Whh0, Wih1, Whh1, bih0, bhh0, bih1, bhh1,
                                        Wr1, frag0, frag1, bias0, bias1, wr1h,
                                        h0buf, h1buf);

  lstm_persist<<<NWG, BLK, 0, stream>>>(x, Wih0, bias0, bias1, frag0, frag1,
                                        h0buf, h1buf, hist, hmean, bar);

  head1_kernel<<<64, 128, 0, stream>>>(hist, hmean, Wl1, b1v, wr1h, m1, elu_l);
  head2_kernel<<<64, 128, 0, stream>>>(m1, elu_l, Wl2, b2v, Wr2, sbuf);
  head3_kernel<<<512, 64, 0, stream>>>(sbuf, Wfc1, bfc1, Wfc2, bfc2, out);
}
